// Controller_85143431676138
// MI455X (gfx1250) — matrix operand order repaired and checked by reading, not compiled, not run on hardware
//
#include <hip/hip_runtime.h>
#include <hip/hip_bf16.h>
#include <stdint.h>

// ---------------------------------------------------------------------------
// ENAS LSTM controller, single persistent workgroup, CDNA5 (gfx1250).
// Gate GEMV (1x128 @ 128x256) on v_wmma_f32_16x16x32_f16.
// B (weight) fragments are loaded ONCE into 64 VGPRs per lane and stay
// register-resident across all ~54 sequential LSTM steps (occupancy-1 kernel,
// so VGPRs are free); per-step work is only the A fragment + 8 WMMAs.
// ---------------------------------------------------------------------------

#define H        64
#define GATES    256          // 4*H
#define KK       128          // concat(x, h)
#define NPAIR    64           // KK/2 packed f16 pairs
#define NUMNODES 7
#define TANH_C   1.1f
#define OP_RED   2.5f
#define NTHREADS 256

typedef __attribute__((ext_vector_type(16))) _Float16 v16h;
typedef __attribute__((ext_vector_type(8)))  float    v8f;

struct __align__(16) Smem {
  uint32_t WcatP[NPAIR * GATES];   // packed f16 K-pairs: [k/2][n], 64 KB
  float Wip[H * H];                // w_index_prev, row-major [out][in]
  float Wic[H * H];                // w_index_curr
  float Wio[H];                    // w_index_out
  float Wops[5 * H];
  float bops[5];
  float abias[5];
  float efirst[H];
  float eops[5 * H];
  float h[H];
  float c[H];
  float xvec[H];                   // current LSTM input vector
  uint32_t xhP[NPAIR];             // packed f16 pairs of [x, h]
  float gates[GATES];
  float lg[H];
  float prev_fc[NUMNODES][H];
  float prev_h[NUMNODES][H];
  float scal[8];
  int   chosen;
};

static_assert(sizeof(Smem) < 160 * 1024, "LDS budget");

__device__ __forceinline__ uint32_t pack2h(float a, float b) {
  union { _Float16 hh[2]; uint32_t u; } cv;
  cv.hh[0] = (_Float16)a;
  cv.hh[1] = (_Float16)b;
  return cv.u;
}

__device__ __forceinline__ float sigm(float x) { return 1.0f / (1.0f + expf(-x)); }

__device__ __forceinline__ uint32_t rotl32(uint32_t x, int n) {
  return (x << n) | (x >> (32 - n));
}

// JAX Threefry-2x32 (20 rounds), matches jax._src.prng.threefry2x32.
__device__ void threefry2x32(uint32_t k0, uint32_t k1, uint32_t x0, uint32_t x1,
                             uint32_t& o0, uint32_t& o1) {
  const uint32_t ks2 = k0 ^ k1 ^ 0x1BD11BDAu;
  uint32_t v0 = x0 + k0;
  uint32_t v1 = x1 + k1;
#define TF_R(r) { v0 += v1; v1 = rotl32(v1, r); v1 ^= v0; }
  TF_R(13) TF_R(15) TF_R(26) TF_R(6)
  v0 += k1;  v1 += ks2 + 1u;
  TF_R(17) TF_R(29) TF_R(16) TF_R(24)
  v0 += ks2; v1 += k0 + 2u;
  TF_R(13) TF_R(15) TF_R(26) TF_R(6)
  v0 += k0;  v1 += k1 + 3u;
  TF_R(17) TF_R(29) TF_R(16) TF_R(24)
  v0 += k1;  v1 += ks2 + 4u;
  TF_R(13) TF_R(15) TF_R(26) TF_R(6)
  v0 += ks2; v1 += k0 + 5u;
#undef TF_R
  o0 = v0; o1 = v1;
}

// jax.random.categorical(fold_in(key(42), step), logits) + log_softmax stats.
// Reproduces threefry_random_bits iota split with odd-size zero padding.
__device__ int sample_cat(const float* logits, int n, uint32_t step,
                          float& lp_sel, float& ent) {
  // fold_in: new key = threefry(key=(0,42), count=(0, step))
  uint32_t fk0, fk1;
  threefry2x32(0u, 42u, 0u, step, fk0, fk1);

  uint32_t bits[8];
  const int nh = (n + 1) >> 1;               // half size after odd padding
  for (int i = 0; i < nh; ++i) {
    uint32_t c0 = (uint32_t)i;
    uint32_t c1 = (nh + i < n) ? (uint32_t)(nh + i) : 0u;
    uint32_t o0, o1;
    threefry2x32(fk0, fk1, c0, c1, o0, o1);
    bits[i] = o0;
    if (nh + i < n) bits[nh + i] = o1;
  }

  // log_softmax
  float m = -1e30f;
  for (int j = 0; j < n; ++j) m = fmaxf(m, logits[j]);
  float sum = 0.0f;
  for (int j = 0; j < n; ++j) sum += expf(logits[j] - m);
  const float lse = m + logf(sum);

  const float TINY = 1.17549435e-38f;
  int   best  = 0;
  float bestz = -1e30f;
  float e     = 0.0f;
  for (int j = 0; j < n; ++j) {
    float lp = logits[j] - lse;
    e -= expf(lp) * lp;
    union { uint32_t u; float f; } cv;
    cv.u = (bits[j] >> 9) | 0x3F800000u;     // [1,2)
    float u = cv.f - 1.0f;                   // [0,1)
    u = u * (1.0f - TINY) + TINY;            // uniform(minval=tiny)
    float g = -logf(-logf(u));               // gumbel
    float z = logits[j] + g;
    if (z > bestz) { bestz = z; best = j; }  // first max, like jnp.argmax
  }
  lp_sel = logits[best] - lse;
  ent    = e;
  return best;
}

// Load this lane's register-resident B fragments (2 N-tiles x 4 K-steps).
// B layout used here (contiguous K, same order as A): lanes 0-15 K=0..15, lanes 16-31 K=16..31,
// VGPR i holds the f16 pair (K=2i, 2i+1) within the half.
__device__ __forceinline__ void load_bfrags(const Smem* s, int tid,
                                            v16h Bf[2][4]) {
  const int lane = tid & 31;
  const int wave = tid >> 5;
  const int half = lane >> 4;
  const int nloc = lane & 15;
#pragma unroll
  for (int t = 0; t < 2; ++t) {
    const int n = (wave * 2 + t) * 16 + nloc;
#pragma unroll
    for (int ks = 0; ks < 4; ++ks) {
      v16h B;
#pragma unroll
      for (int i = 0; i < 8; ++i) {
        const int pi = ks * 16 + half * 8 + i;      // global K-pair index
        union { uint32_t u; _Float16 hh[2]; } cv;
        cv.u = s->WcatP[pi * GATES + n];
        B[2 * i]     = cv.hh[0];
        B[2 * i + 1] = cv.hh[1];
      }
      Bf[t][ks] = B;
    }
  }
}

// One LSTM cell step: gates = [x,h] @ Wcat via WMMA, then state update.
__device__ __forceinline__ void lstm_cell(Smem* s, int tid,
                                          const v16h (&Bf)[2][4]) {
  const int lane = tid & 31;
  const int wave = tid >> 5;

  // Pack [xvec, h] into 64 f16 pairs.
  if (tid < NPAIR) {
    float e0, e1;
    if (tid < 32) { e0 = s->xvec[2 * tid];        e1 = s->xvec[2 * tid + 1]; }
    else          { int t = tid - 32; e0 = s->h[2 * t]; e1 = s->h[2 * t + 1]; }
    s->xhP[tid] = pack2h(e0, e1);
  }
  __syncthreads();

  const int half = lane >> 4;       // K-half of the wave
  const int nloc = lane & 15;       // N within tile / M row for A

  // A fragments (16x32 f16), all 4 K-steps, branchless: every lane in a
  // half-wave reads the same xhP dword (LDS broadcast); rows M!=0 are zeroed
  // via cndmask instead of EXEC branching.
  v16h A[4];
#pragma unroll
  for (int ks = 0; ks < 4; ++ks) {
    v16h a = {};
#pragma unroll
    for (int i = 0; i < 8; ++i) {
      // VGPR i holds K = base..base+1; base/2 = 8*half + i
      const int pl = 8 * half + i;
      uint32_t p = s->xhP[ks * 16 + pl];
      p = (nloc == 0) ? p : 0u;               // only row M=0 is real
      union { uint32_t u; _Float16 hh[2]; } cv;
      cv.u = p;
      a[2 * i]     = cv.hh[0];
      a[2 * i + 1] = cv.hh[1];
    }
    A[ks] = a;
  }

  // 8 back-to-back WMMAs against register-resident B.
  v8f acc0 = {};
  v8f acc1 = {};
#pragma unroll
  for (int ks = 0; ks < 4; ++ks) {
    acc0 = __builtin_amdgcn_wmma_f32_16x16x32_f16(false, A[ks], false, Bf[0][ks],
                                                  (short)0, acc0, false, false);
    acc1 = __builtin_amdgcn_wmma_f32_16x16x32_f16(false, A[ks], false, Bf[1][ks],
                                                  (short)0, acc1, false, false);
  }

  // D row M=0 lives in C-VGPR0 of lanes 0-15 (N = lane).
  if (lane < 16) {
    s->gates[(wave * 2) * 16 + lane]     = acc0[0];
    s->gates[(wave * 2 + 1) * 16 + lane] = acc1[0];
  }
  __syncthreads();

  if (tid < H) {
    float ig = s->gates[tid];
    float fg = s->gates[H + tid];
    float gg = s->gates[2 * H + tid];
    float og = s->gates[3 * H + tid];
    float cc = sigm(fg) * s->c[tid] + sigm(ig) * tanhf(gg);
    float hh = sigm(og) * tanhf(cc);
    s->c[tid] = cc;
    s->h[tid] = hh;
  }
  __syncthreads();
}

__global__ __launch_bounds__(NTHREADS, 1)
void enas_controller_kernel(const float* __restrict__ embed_first,
                            const float* __restrict__ embed_ops,
                            const float* __restrict__ w_ih,
                            const float* __restrict__ w_hh,
                            const float* __restrict__ w_index_prev,
                            const float* __restrict__ w_index_curr,
                            const float* __restrict__ w_index_out,
                            const float* __restrict__ w_ops,
                            const float* __restrict__ b_ops,
                            const float* __restrict__ add_bias,
                            int*   __restrict__ out_arc,   // 40 int32
                            float* __restrict__ out_ent,   // 40 f32
                            float* __restrict__ out_lp) {  // 40 f32
  extern __shared__ __align__(16) char smem_raw[];
  Smem* s = reinterpret_cast<Smem*>(smem_raw);
  const int tid = threadIdx.x;

  if (tid == 0) {                    // global_prefetch_b8 hints
    __builtin_prefetch(w_ih, 0, 3);
    __builtin_prefetch(w_hh, 0, 3);
    __builtin_prefetch(w_index_prev, 0, 3);
    __builtin_prefetch(w_index_curr, 0, 3);
  }

  // ---- Stage all weights into LDS (one-time HBM/L2 traffic) ----
  // Wcat[k][n]: k<64 -> w_ih[n][k]; k>=64 -> w_hh[n][k-64]. Packed K-pairs.
  for (int idx = tid; idx < NPAIR * GATES; idx += NTHREADS) {
    int pk = idx >> 8;               // K-pair 0..63 (k = 2*pk even -> no straddle)
    int n  = idx & 255;
    int k0 = 2 * pk;
    float e0 = (k0 < H) ? w_ih[n * H + k0]     : w_hh[n * H + (k0 - H)];
    float e1 = (k0 < H) ? w_ih[n * H + k0 + 1] : w_hh[n * H + (k0 + 1 - H)];
    s->WcatP[pk * GATES + n] = pack2h(e0, e1);
  }
  for (int idx = tid; idx < H * H; idx += NTHREADS) {
    s->Wip[idx] = w_index_prev[idx];
    s->Wic[idx] = w_index_curr[idx];
  }
  for (int idx = tid; idx < 5 * H; idx += NTHREADS) {
    s->Wops[idx] = w_ops[idx];
    s->eops[idx] = embed_ops[idx];
  }
  if (tid < H) {
    s->Wio[tid]    = w_index_out[tid];
    s->efirst[tid] = embed_first[tid];
    s->h[tid] = 0.0f;
    s->c[tid] = 0.0f;
  }
  if (tid < 5) {
    s->bops[tid]  = b_ops[tid];
    s->abias[tid] = add_bias[tid];
  }
  __syncthreads();

  // ---- One-time: pull this lane's B fragments into resident VGPRs ----
  v16h Bf[2][4];
  load_bfrags(s, tid, Bf);

  uint32_t rng_step = 0;
  int outpos = 0;

  for (int cell = 0; cell < 2; ++cell) {     // normal + reduction cell
    if (tid < H) s->xvec[tid] = s->efirst[tid];
    __syncthreads();

    // Warmup: 2 cells; prev_h = zeros, prev_fc = h @ Wip^T
    for (int w = 0; w < 2; ++w) {
      lstm_cell(s, tid, Bf);
      if (tid < H) {
        s->prev_h[w][tid] = 0.0f;
        float acc = 0.0f;
#pragma unroll 8
        for (int k = 0; k < H; ++k) acc += s->h[k] * s->Wip[tid * H + k];
        s->prev_fc[w][tid] = acc;
      }
      __syncthreads();
    }

    for (int node = 2; node < NUMNODES; ++node) {
      // ---- two index (attention) samples ----
      for (int r = 0; r < 2; ++r) {
        lstm_cell(s, tid, Bf);
        if (tid < H) {                        // lg = h @ Wic^T
          float acc = 0.0f;
#pragma unroll 8
          for (int k = 0; k < H; ++k) acc += s->h[k] * s->Wic[tid * H + k];
          s->lg[tid] = acc;
        }
        __syncthreads();
        if (tid < node) {                     // q_j = tanh(fc_j + lg) . Wio
          float acc = 0.0f;
          for (int k = 0; k < H; ++k)
            acc += tanhf(s->prev_fc[tid][k] + s->lg[k]) * s->Wio[k];
          s->scal[tid] = acc;
        }
        __syncthreads();
        ++rng_step;
        if (tid == 0) {
          float logits[8];
          for (int j = 0; j < node; ++j) logits[j] = TANH_C * tanhf(s->scal[j]);
          float lp, ent;
          int a = sample_cat(logits, node, rng_step, lp, ent);
          out_arc[outpos] = a;
          out_ent[outpos] = ent;
          out_lp[outpos]  = lp;
          s->chosen = a;
        }
        __syncthreads();
        if (tid < H) s->xvec[tid] = s->prev_h[s->chosen][tid];
        __syncthreads();
        ++outpos;
      }
      // ---- two op samples ----
      for (int r = 0; r < 2; ++r) {
        lstm_cell(s, tid, Bf);
        if (tid < 5) {
          float acc = 0.0f;
          for (int k = 0; k < H; ++k) acc += s->h[k] * s->Wops[tid * H + k];
          s->scal[tid] = acc + s->bops[tid];
        }
        __syncthreads();
        ++rng_step;
        if (tid == 0) {
          float logits[8];
          for (int j = 0; j < 5; ++j)
            logits[j] = (TANH_C / OP_RED) * tanhf(s->scal[j]) + s->abias[j];
          float lp, ent;
          int a = sample_cat(logits, 5, rng_step, lp, ent);
          out_arc[outpos] = a;
          out_ent[outpos] = ent;
          out_lp[outpos]  = lp;
          s->chosen = a;
        }
        __syncthreads();
        if (tid < H) s->xvec[tid] = s->eops[s->chosen * H + tid];
        __syncthreads();
        ++outpos;
      }
      // ---- node finalize ----
      lstm_cell(s, tid, Bf);
      if (tid < H) {
        s->prev_h[node][tid] = s->h[tid];
        float acc = 0.0f;
#pragma unroll 8
        for (int k = 0; k < H; ++k) acc += s->h[k] * s->Wip[tid * H + k];
        s->prev_fc[node][tid] = acc;
        s->xvec[tid] = s->efirst[tid];
      }
      __syncthreads();
    }
  }
}

extern "C" void kernel_launch(void* const* d_in, const int* in_sizes, int n_in,
                              void* d_out, int out_size, void* d_ws, size_t ws_size,
                              hipStream_t stream) {
  (void)in_sizes; (void)n_in; (void)out_size; (void)d_ws; (void)ws_size;
  const float* embed_first  = (const float*)d_in[0];
  const float* embed_ops    = (const float*)d_in[1];
  const float* w_ih         = (const float*)d_in[2];
  const float* w_hh         = (const float*)d_in[3];
  const float* w_index_prev = (const float*)d_in[4];
  const float* w_index_curr = (const float*)d_in[5];
  const float* w_index_out  = (const float*)d_in[6];
  const float* w_ops        = (const float*)d_in[7];
  const float* b_ops        = (const float*)d_in[8];
  const float* add_bias     = (const float*)d_in[9];

  // d_out layout: [0..39] arcs (int32: normal 0-19, reduction 20-39),
  //               [40..79] entropies (f32), [80..119] log-probs (f32).
  int*   out_arc = (int*)d_out;
  float* out_f   = (float*)d_out;

  enas_controller_kernel<<<1, NTHREADS, sizeof(Smem), stream>>>(
      embed_first, embed_ops, w_ih, w_hh, w_index_prev, w_index_curr,
      w_index_out, w_ops, b_ops, add_bias,
      out_arc, out_f + 40, out_f + 80);
}
